// MultiHeadAttention_29033978921231
// MI455X (gfx1250) — matrix operand order repaired and checked by reading, not compiled, not run on hardware
//
#include <hip/hip_runtime.h>

// ---------------------------------------------------------------------------
// MI455X (gfx1250) fused multi-head attention, bf16 WMMA path.
//   B=2, L=2048, D_MODEL=1024, H=16, DH=64, INNER=1024
// Stages (all on `stream`):
//   0) pack_bf16_kernel : one-time f32 -> bf16 conversion of q,k,v,Wq,Wk,Wv,Wo
//   1) qkv_proj_kernel  : [q|k|v]@W^T -> Qh/Kh ([B,H,L,DH]) and Vt ([B,H,DH,L])
//                         (double-buffered GLOBAL_LOAD_ASYNC_TO_LDS staging)
//   2) rope_kernel      : in-place RoPE on Qh, Kh
//   3) attn_kernel      : flash attention with online softmax, bf16 WMMA
//   4) out_proj_kernel  : O @ Wo^T -> f32 d_out
// ---------------------------------------------------------------------------

typedef __attribute__((ext_vector_type(16))) __bf16 v16bf;
typedef __attribute__((ext_vector_type(8)))  __bf16 v8bf;
typedef __attribute__((ext_vector_type(4)))  __bf16 v4bf;
typedef __attribute__((ext_vector_type(8)))  float  v8f;
typedef __attribute__((ext_vector_type(4)))  int    v4i;

#define B_      2
#define L_      2048
#define DM_     1024
#define H_      16
#define DH_     64
#define INNER_  1024
#define LDT     72   // padded LDS leading dim (bf16 elems) to dodge bank conflicts

// ---- gfx1250 async-copy-to-LDS path (guarded; falls back to sync loads) ----
#if defined(__gfx1250__) && __has_builtin(__builtin_amdgcn_global_load_async_to_lds_b128) && \
    __has_builtin(__builtin_amdgcn_s_wait_asynccnt)
#define USE_ASYNC_LDS 1
#else
#define USE_ASYNC_LDS 0
#endif

#if USE_ASYNC_LDS
typedef __attribute__((address_space(1))) v4i as1_v4i;   // prints as "int4 __device__ *"
typedef __attribute__((address_space(3))) v4i as3_v4i;   // prints as "int4 __shared__ *"
#define AS1P(p) ((as1_v4i*)(unsigned long long)(const void*)(p))
#define AS3P(p) ((as3_v4i*)(unsigned int)(unsigned long long)(const void*)(p))
#define ASYNC_WAIT(n) __builtin_amdgcn_s_wait_asynccnt(n)
#else
#define ASYNC_WAIT(n)
#endif

__device__ __forceinline__ v8f wmma_bf16(v16bf a, v16bf b, v8f c) {
  // v_wmma_f32_16x16x32_bf16  D = A(16x32) * B(32x16) + C(16x16 f32)
  return __builtin_amdgcn_wmma_f32_16x16x32_bf16(false, a, false, b, (short)0, c, false, false);
}

__device__ __forceinline__ v8f vzero8() {
  v8f z = {0.f,0.f,0.f,0.f,0.f,0.f,0.f,0.f};
  return z;
}

// A-matrix frag (16x32 bf16, row-major source): lane<16 -> M=lane, K{0..7,16..23};
// lane>=16 -> M=lane-16, K{8..15,24..31}
__device__ __forceinline__ v16bf load_a_frag(const __bf16* base, int ld) {
  const int lane = threadIdx.x & 31;
  const int row  = lane & 15;
  const int koff = (lane >> 4) * 8;
  const __bf16* p = base + row * ld + koff;
  union { v16bf v; v8bf h[2]; } u;
  u.h[0] = *(const v8bf*)(p);
  u.h[1] = *(const v8bf*)(p + 16);
  return u.v;
}

// B-matrix frag (32x16 bf16). Source rows = N (provider rows: W rows / K rows / Vt rows),
// lane<16 -> N=lane, K{0..7,16..23}; lane>=16 -> N=lane-16, K{8..15,24..31}.
__device__ __forceinline__ v16bf load_b_frag(const __bf16* base, int ld) {
  const int lane = threadIdx.x & 31;
  const int n    = lane & 15;
  const int koff = (lane >> 4) * 8;
  const __bf16* p = base + n * ld + koff;
  union { v16bf v; v8bf h[2]; } u;
  u.h[0] = *(const v8bf*)(p);
  u.h[1] = *(const v8bf*)(p + 16);
  return u.v;
}

// Stage one 128x64 bf16 tile pair (A rows m-block, B rows n-block; both K-stride 1024)
// into LDS. Async path: 8 GLOBAL_LOAD_ASYNC_TO_LDS_B128 per thread per tile pair.
__device__ __forceinline__ void stage_pair(const __bf16* __restrict__ Agb,
                                           const __bf16* __restrict__ Wgb, int k0,
                                           __bf16* Al, __bf16* Bl, int tid) {
#pragma unroll
  for (int it = 0; it < 4; ++it) {
    int chunk = tid + it * 256;
    int row = chunk >> 3;          // 0..127
    int c8  = (chunk & 7) * 8;     // 0..56
    const __bf16* ga = Agb + (size_t)row * 1024 + k0 + c8;
    const __bf16* gb = Wgb + (size_t)row * 1024 + k0 + c8;
    __bf16* la = Al + row * LDT + c8;
    __bf16* lb = Bl + row * LDT + c8;
#if USE_ASYNC_LDS
    __builtin_amdgcn_global_load_async_to_lds_b128(AS1P(ga), AS3P(la), 0, 0);
    __builtin_amdgcn_global_load_async_to_lds_b128(AS1P(gb), AS3P(lb), 0, 0);
#else
    *(v8bf*)la = *(const v8bf*)ga;
    *(v8bf*)lb = *(const v8bf*)gb;
#endif
  }
}

// ---------------------------------------------------------------------------
// Kernel 0: f32 -> bf16 pack (streaming; one-time).
// ---------------------------------------------------------------------------
__global__ __launch_bounds__(256) void pack_bf16_kernel(const float* __restrict__ src,
                                                        __bf16* __restrict__ dst, int n) {
  int i = (blockIdx.x * 256 + threadIdx.x) * 4;
  if (i >= n) return;
  float4 f = *(const float4*)(src + i);
  v4bf o; o[0] = (__bf16)f.x; o[1] = (__bf16)f.y; o[2] = (__bf16)f.z; o[3] = (__bf16)f.w;
  *(v4bf*)(dst + i) = o;
}

// ---------------------------------------------------------------------------
// Kernel 1: fused QKV projection.  grid = (M/128, 3*1024/128) = (32, 24)
// Block tile 128x128, 8 waves in 4(M) x 2(N), each wave -> 32x64 (8 WMMAs / K-step).
// Double-buffered async tile staging.
// ---------------------------------------------------------------------------
__global__ __launch_bounds__(256) void qkv_proj_kernel(
    const __bf16* __restrict__ qb, const __bf16* __restrict__ kb, const __bf16* __restrict__ vb,
    const __bf16* __restrict__ Wqb, const __bf16* __restrict__ Wkb, const __bf16* __restrict__ Wvb,
    __bf16* __restrict__ Qh, __bf16* __restrict__ Kh, __bf16* __restrict__ Vt)
{
  __shared__ __bf16 Al[2][128 * LDT];
  __shared__ __bf16 Bl[2][128 * LDT];

  const int tid  = threadIdx.x;
  const int lane = tid & 31;
  const int w    = tid >> 5;
  const int wm   = w >> 1;            // 0..3
  const int wn   = w & 1;             // 0..1
  const int half = lane >> 4;
  const int c    = lane & 15;

  const int m0    = blockIdx.x * 128;
  const int which = blockIdx.y >> 3;  // 0=q,1=k,2=v
  const int n0    = (blockIdx.y & 7) * 128;

  const __bf16* Agb = ((which == 0) ? qb  : (which == 1) ? kb  : vb)  + (size_t)m0 * DM_;
  const __bf16* Wgb = ((which == 0) ? Wqb : (which == 1) ? Wkb : Wvb) + (size_t)n0 * DM_;

  v8f acc[2][4];
#pragma unroll
  for (int sm = 0; sm < 2; ++sm)
#pragma unroll
    for (int sn = 0; sn < 4; ++sn) acc[sm][sn] = vzero8();

  stage_pair(Agb, Wgb, 0, Al[0], Bl[0], tid);

  for (int t = 0; t < 16; ++t) {
    const int cur = t & 1;
    if (t < 15) {
      stage_pair(Agb, Wgb, (t + 1) * 64, Al[1 - cur], Bl[1 - cur], tid);
      ASYNC_WAIT(8);   // current tile's 8 async ops done; next tile's 8 in flight
    } else {
      ASYNC_WAIT(0);
    }
    __syncthreads();

#pragma unroll
    for (int s = 0; s < 2; ++s) {
      v16bf af[2], bfr[4];
#pragma unroll
      for (int sm = 0; sm < 2; ++sm)
        af[sm] = load_a_frag(&Al[cur][(wm * 32 + sm * 16) * LDT + s * 32], LDT);
#pragma unroll
      for (int sn = 0; sn < 4; ++sn)
        bfr[sn] = load_b_frag(&Bl[cur][(wn * 64 + sn * 16) * LDT + s * 32], LDT);
#pragma unroll
      for (int sm = 0; sm < 2; ++sm)
#pragma unroll
        for (int sn = 0; sn < 4; ++sn)
          acc[sm][sn] = wmma_bf16(af[sm], bfr[sn], acc[sm][sn]);
    }
    __syncthreads();   // all waves done reading buf[cur] before it is restaged
  }

  // scatter bf16 results into head layouts
#pragma unroll
  for (int sm = 0; sm < 2; ++sm)
#pragma unroll
    for (int sn = 0; sn < 4; ++sn)
#pragma unroll
      for (int r = 0; r < 8; ++r) {
        int m    = m0 + wm * 32 + sm * 16 + r + 8 * half;
        int nloc = n0 + wn * 64 + sn * 16 + c;
        int b    = m >> 11;
        int pos  = m & (L_ - 1);
        int h    = nloc >> 6;
        int d    = nloc & 63;
        __bf16 val = (__bf16)acc[sm][sn][r];
        if (which == 2)
          Vt[(((size_t)b * H_ + h) * DH_ + d) * L_ + pos] = val;        // [B,H,DH,L]
        else {
          __bf16* dst = (which == 0) ? Qh : Kh;
          dst[(((size_t)b * H_ + h) * L_ + pos) * DH_ + d] = val;       // [B,H,L,DH]
        }
      }
}

// ---------------------------------------------------------------------------
// Kernel 2: RoPE in-place on Qh, Kh.  One thread per (b,h,pos,i<32) pair.
// ---------------------------------------------------------------------------
__global__ __launch_bounds__(256) void rope_kernel(__bf16* __restrict__ Qh,
                                                   __bf16* __restrict__ Kh)
{
  int idx = blockIdx.x * blockDim.x + threadIdx.x;
  if (idx >= B_ * H_ * L_ * (DH_ / 2)) return;
  int i   = idx & 31;
  int t   = idx >> 5;
  int pos = t & (L_ - 1);
  int bh  = t >> 11;
  float inv_freq = __expf(-(float)(2 * i) * (9.2103403719761836f / 64.0f));
  float ang = (float)pos * inv_freq;
  float s, cco;
  __sincosf(ang, &s, &cco);
  size_t base = ((size_t)bh * L_ + pos) * DH_;

  float q1 = (float)Qh[base + i], q2 = (float)Qh[base + i + 32];
  Qh[base + i]      = (__bf16)(q1 * cco - q2 * s);
  Qh[base + i + 32] = (__bf16)(q2 * cco + q1 * s);

  float k1 = (float)Kh[base + i], k2 = (float)Kh[base + i + 32];
  Kh[base + i]      = (__bf16)(k1 * cco - k2 * s);
  Kh[base + i + 32] = (__bf16)(k2 * cco + k1 * s);
}

// ---------------------------------------------------------------------------
// Kernel 3: flash attention.  grid = (L/128, B*H) = (16, 32), 256 threads.
// Each wave owns 16 query rows; streams keys in blocks of 64 with online softmax.
// ---------------------------------------------------------------------------
__global__ __launch_bounds__(256) void attn_kernel(
    const __bf16* __restrict__ Qh, const __bf16* __restrict__ Kh,
    const __bf16* __restrict__ Vt, const unsigned char* __restrict__ mask,
    __bf16* __restrict__ Oa)
{
  __shared__ __bf16 Plds[8 * 16 * LDT];   // per-wave 16x64 P restage buffer

  const int tid  = threadIdx.x;
  const int lane = tid & 31;
  const int w    = tid >> 5;
  const int half = lane >> 4;
  const int c    = lane & 15;

  const int bh = blockIdx.y;
  const int b  = bh >> 4;
  const int h  = bh & 15;
  const int q0 = blockIdx.x * 128 + w * 16;

  const __bf16* Qp = Qh + (size_t)bh * L_ * DH_;
  const __bf16* Kp = Kh + (size_t)bh * L_ * DH_;
  const __bf16* Vp = Vt + (size_t)bh * DH_ * L_;
  const unsigned char* mp = mask + (size_t)b * L_;

  const float scale = 0.125f;           // 1/sqrt(64)

  v16bf aq0 = load_a_frag(Qp + (size_t)q0 * DH_,      DH_);
  v16bf aq1 = load_a_frag(Qp + (size_t)q0 * DH_ + 32, DH_);

  v8f O[4];
#pragma unroll
  for (int i = 0; i < 4; ++i) O[i] = vzero8();
  float mrow[8], lrow[8];
#pragma unroll
  for (int r = 0; r < 8; ++r) { mrow[r] = -1e30f; lrow[r] = 0.f; }

  __bf16* pl = &Plds[w * 16 * LDT];

  for (int kbk = 0; kbk < L_; kbk += 64) {
    // ---- S = scale * Q K^T + bias, four 16x16 key subtiles ----
    v8f S[4];
#pragma unroll
    for (int j = 0; j < 4; ++j) {
      v16bf bk0 = load_b_frag(Kp + (size_t)(kbk + 16 * j) * DH_,      DH_);
      v16bf bk1 = load_b_frag(Kp + (size_t)(kbk + 16 * j) * DH_ + 32, DH_);
      S[j] = wmma_bf16(aq0, bk0, vzero8());
      S[j] = wmma_bf16(aq1, bk1, S[j]);
    }
    float bias[4];
#pragma unroll
    for (int j = 0; j < 4; ++j)
      bias[j] = mp[kbk + 16 * j + c] ? 0.f : -1e30f;
#pragma unroll
    for (int j = 0; j < 4; ++j)
#pragma unroll
      for (int r = 0; r < 8; ++r)
        S[j][r] = S[j][r] * scale + bias[j];

    // ---- online softmax update (16-lane row reductions, wave32) ----
#pragma unroll
    for (int r = 0; r < 8; ++r) {
      float mx = fmaxf(fmaxf(S[0][r], S[1][r]), fmaxf(S[2][r], S[3][r]));
#pragma unroll
      for (int off = 1; off < 16; off <<= 1)
        mx = fmaxf(mx, __shfl_xor(mx, off, 32));
      float mnew  = fmaxf(mrow[r], mx);
      float alpha = __expf(mrow[r] - mnew);
      float rsum  = 0.f;
#pragma unroll
      for (int j = 0; j < 4; ++j) {
        float e = __expf(S[j][r] - mnew);
        S[j][r] = e;
        rsum += e;
      }
#pragma unroll
      for (int off = 1; off < 16; off <<= 1)
        rsum += __shfl_xor(rsum, off, 32);
      lrow[r] = lrow[r] * alpha + rsum;
      mrow[r] = mnew;
#pragma unroll
      for (int i = 0; i < 4; ++i) O[i][r] *= alpha;
    }

    // ---- restage P (C-layout f32) -> LDS -> A-layout bf16 frags ----
#pragma unroll
    for (int j = 0; j < 4; ++j)
#pragma unroll
      for (int r = 0; r < 8; ++r)
        pl[(r + 8 * half) * LDT + 16 * j + c] = (__bf16)S[j][r];
    asm volatile("s_wait_dscnt 0" ::: "memory");
    v16bf p0 = load_a_frag(pl,      LDT);   // keys kbk+0..31
    v16bf p1 = load_a_frag(pl + 32, LDT);   // keys kbk+32..63

    // ---- O += P @ V  (Vt rows are d, contiguous keys) ----
#pragma unroll
    for (int i = 0; i < 4; ++i) {
      v16bf bv0 = load_b_frag(Vp + (size_t)(16 * i) * L_ + kbk,      L_);
      v16bf bv1 = load_b_frag(Vp + (size_t)(16 * i) * L_ + kbk + 32, L_);
      O[i] = wmma_bf16(p0, bv0, O[i]);
      O[i] = wmma_bf16(p1, bv1, O[i]);
    }
  }

  // ---- normalize and store to [B, L, INNER] bf16 ----
#pragma unroll
  for (int r = 0; r < 8; ++r) {
    float inv = 1.0f / lrow[r];
    int pos = q0 + r + 8 * half;
#pragma unroll
    for (int i = 0; i < 4; ++i)
      Oa[((size_t)b * L_ + pos) * INNER_ + h * DH_ + 16 * i + c] =
          (__bf16)(O[i][r] * inv);
  }
}

// ---------------------------------------------------------------------------
// Kernel 4: output projection O @ Wo^T -> f32.  grid = (32, 8)
// ---------------------------------------------------------------------------
__global__ __launch_bounds__(256) void out_proj_kernel(
    const __bf16* __restrict__ Oa, const __bf16* __restrict__ Wob,
    float* __restrict__ out)
{
  __shared__ __bf16 Al[2][128 * LDT];
  __shared__ __bf16 Bl[2][128 * LDT];

  const int tid  = threadIdx.x;
  const int lane = tid & 31;
  const int w    = tid >> 5;
  const int wm   = w >> 1;
  const int wn   = w & 1;
  const int half = lane >> 4;
  const int c    = lane & 15;

  const int m0 = blockIdx.x * 128;
  const int n0 = blockIdx.y * 128;

  const __bf16* Agb = Oa  + (size_t)m0 * INNER_;
  const __bf16* Wgb = Wob + (size_t)n0 * INNER_;

  v8f acc[2][4];
#pragma unroll
  for (int sm = 0; sm < 2; ++sm)
#pragma unroll
    for (int sn = 0; sn < 4; ++sn) acc[sm][sn] = vzero8();

  stage_pair(Agb, Wgb, 0, Al[0], Bl[0], tid);

  for (int t = 0; t < 16; ++t) {
    const int cur = t & 1;
    if (t < 15) {
      stage_pair(Agb, Wgb, (t + 1) * 64, Al[1 - cur], Bl[1 - cur], tid);
      ASYNC_WAIT(8);
    } else {
      ASYNC_WAIT(0);
    }
    __syncthreads();

#pragma unroll
    for (int s = 0; s < 2; ++s) {
      v16bf af[2], bfr[4];
#pragma unroll
      for (int sm = 0; sm < 2; ++sm)
        af[sm] = load_a_frag(&Al[cur][(wm * 32 + sm * 16) * LDT + s * 32], LDT);
#pragma unroll
      for (int sn = 0; sn < 4; ++sn)
        bfr[sn] = load_b_frag(&Bl[cur][(wn * 64 + sn * 16) * LDT + s * 32], LDT);
#pragma unroll
      for (int sm = 0; sm < 2; ++sm)
#pragma unroll
        for (int sn = 0; sn < 4; ++sn)
          acc[sm][sn] = wmma_bf16(af[sm], bfr[sn], acc[sm][sn]);
    }
    __syncthreads();
  }

#pragma unroll
  for (int sm = 0; sm < 2; ++sm)
#pragma unroll
    for (int sn = 0; sn < 4; ++sn)
#pragma unroll
      for (int r = 0; r < 8; ++r) {
        int m = m0 + wm * 32 + sm * 16 + r + 8 * half;
        int n = n0 + wn * 64 + sn * 16 + c;
        out[(size_t)m * DM_ + n] = acc[sm][sn][r];
      }
}

// ---------------------------------------------------------------------------
extern "C" void kernel_launch(void* const* d_in, const int* in_sizes, int n_in,
                              void* d_out, int out_size, void* d_ws, size_t ws_size,
                              hipStream_t stream) {
  const float* q  = (const float*)d_in[0];
  const float* k  = (const float*)d_in[1];
  const float* v  = (const float*)d_in[2];
  const unsigned char* mask = (const unsigned char*)d_in[3];
  const float* Wq = (const float*)d_in[4];
  const float* Wk = (const float*)d_in[5];
  const float* Wv = (const float*)d_in[6];
  const float* Wo = (const float*)d_in[7];
  float* out = (float*)d_out;

  const int NQ = B_ * L_ * DM_;     // 4,194,304
  const int NW = INNER_ * DM_;      // 1,048,576

  __bf16* qb  = (__bf16*)d_ws;      // ws layout: 64 MB of bf16 buffers
  __bf16* kbp = qb  + NQ;
  __bf16* vbp = kbp + NQ;
  __bf16* Wqb = vbp + NQ;
  __bf16* Wkb = Wqb + NW;
  __bf16* Wvb = Wkb + NW;
  __bf16* Wob = Wvb + NW;
  __bf16* Qh  = Wob + NW;
  __bf16* Kh  = Qh  + NQ;
  __bf16* Vt  = Kh  + NQ;
  __bf16* Oa  = Vt  + NQ;

  pack_bf16_kernel<<<NQ / 1024, 256, 0, stream>>>(q,  qb,  NQ);
  pack_bf16_kernel<<<NQ / 1024, 256, 0, stream>>>(k,  kbp, NQ);
  pack_bf16_kernel<<<NQ / 1024, 256, 0, stream>>>(v,  vbp, NQ);
  pack_bf16_kernel<<<NW / 1024, 256, 0, stream>>>(Wq, Wqb, NW);
  pack_bf16_kernel<<<NW / 1024, 256, 0, stream>>>(Wk, Wkb, NW);
  pack_bf16_kernel<<<NW / 1024, 256, 0, stream>>>(Wv, Wvb, NW);
  pack_bf16_kernel<<<NW / 1024, 256, 0, stream>>>(Wo, Wob, NW);

  qkv_proj_kernel<<<dim3(32, 24), 256, 0, stream>>>(qb, kbp, vbp, Wqb, Wkb, Wvb, Qh, Kh, Vt);
  rope_kernel<<<(B_ * H_ * L_ * (DH_ / 2)) / 256, 256, 0, stream>>>(Qh, Kh);
  attn_kernel<<<dim3(L_ / 128, B_ * H_), 256, 0, stream>>>(Qh, Kh, Vt, mask, Oa);
  out_proj_kernel<<<dim3(32, 8), 256, 0, stream>>>(Oa, Wob, out);
}
